// SE3EquivariantAttention_75892072120803
// MI455X (gfx1250) — matrix operand order repaired and checked by reading, not compiled, not run on hardware
//
#include <hip/hip_runtime.h>

// ---------------------------------------------------------------------------
// SE(3)-equivariant attention for MI455X (gfx1250), wave32 + WMMA.
// Sizes (fixed by reference): MUL0=16, MUL1=8, P0=32, P1=16, NB=10,
// feat dim = 40, E = 98304, N = 8192, k/v MLP: 10->32->576, p MLP: 10->64->1152
// (pruned to the 72 columns that feed psi = pot[:, :3]).
// ---------------------------------------------------------------------------

typedef __attribute__((ext_vector_type(16))) _Float16 v16h;
typedef __attribute__((ext_vector_type(8)))  float    v8f;

#define INV_SQRT3 0.5773502691896258f
#define SQRT3     1.7320508075688772f
#define NORM24    0.2041241452319315f   // 1/sqrt(MUL0+MUL1)
#define EMB_C     26.669291f            // 1.14136*e^2*sqrt(10)

// Fast transcendental paths: v_exp_f32 / v_rcp_f32 (~1 ulp each). Every
// consumer is an f16 cast, a smooth radial embedding, or a normalized softmax
// ratio, so the precise exp/div expansions would only add VALU latency in the
// shadow of the WMMA stream.
__device__ __forceinline__ float fexp(float x) { return __expf(x); }
__device__ __forceinline__ float frcp(float x) { return __builtin_amdgcn_rcpf(x); }
__device__ __forceinline__ float silu_f(float x) { return x * frcp(1.0f + fexp(-x)); }
__device__ __forceinline__ float sus_f(float x)  {
  return x > 0.0f ? fexp(-frcp(x)) : 0.0f;
}

__device__ __forceinline__ v8f wmma16(v16h a, v16h b, v8f c) {
  // D = A(16x32 f16) * B(32x16 f16) + C(16x16 f32)
  return __builtin_amdgcn_wmma_f32_16x16x32_f16(false, a, false, b, (short)0, c, false, false);
}

// A operand from per-edge padded embedding row (16 halves, K>=16 are zero).
// lane pattern: m = lane&15 selects the row; element j -> K=(lane>>4)*16+j,
// so half 1 (K>=16) is always zero for the 10-dim embedding.
__device__ __forceinline__ v16h load_a_emb(const _Float16* __restrict__ embrow, int half) {
  v16h r;
#pragma unroll
  for (int j = 0; j < 16; ++j) r[j] = (_Float16)0.0f;
#pragma unroll
  for (int j = 0; j < 16; ++j) r[j] = half ? (_Float16)0.0f : embrow[j];
  return r;
}

// A operand from a contiguous 32-wide K slice (LDS hidden activations).
// element j -> K = 2*kb + j, kb = (lane>>4)*8
__device__ __forceinline__ v16h load_a_slice32(const _Float16* __restrict__ p, int kb) {
  v16h r;
#pragma unroll
  for (int j = 0; j < 8; ++j)  r[j] = p[2 * kb + j];
#pragma unroll
  for (int j = 8; j < 16; ++j) r[j] = p[2 * kb + j];
  return r;
}

// B operand: 16 contiguous halves of one column (weights stored transposed,
// [col][K]); lane covers K = (lane>>4)*16 .. +15 (caller applies that offset).
__device__ __forceinline__ v16h load_b16(const _Float16* __restrict__ p) {
  v16h r;
#pragma unroll
  for (int j = 0; j < 16; ++j) r[j] = p[j];
  return r;
}

// ---------------------------------------------------------------------------
// K0: convert MLP weights to f16, transposed into B-operand friendly layout.
//   W1kT [32 cols][32 K] (K rows 10..31 zeroed), W2kT [576][32], same for v,
//   W1pT [64][32], W2pcT [80][64] = the 72 fcp_W2 columns feeding psi (+pad).
// ---------------------------------------------------------------------------
#define OW1K  0
#define OW2K  1024
#define OW1V  19456
#define OW2V  20480
#define OW1P  38912
#define OW2PC 40960
#define WTOT  46080

__global__ void prep_weights_kernel(const float* __restrict__ fckW1, const float* __restrict__ fckW2,
                                    const float* __restrict__ fcvW1, const float* __restrict__ fcvW2,
                                    const float* __restrict__ fcpW1, const float* __restrict__ fcpW2,
                                    _Float16* __restrict__ wc) {
  int tid = blockIdx.x * blockDim.x + threadIdx.x;
  if (tid >= WTOT) return;
  float v = 0.0f;
  if (tid < OW2K) {                       // W1kT
    int col = tid >> 5, k = tid & 31;
    v = (k < 10) ? fckW1[k * 32 + col] : 0.0f;
  } else if (tid < OW1V) {                // W2kT
    int idx = tid - OW2K; int col = idx >> 5, k = idx & 31;
    v = fckW2[k * 576 + col];
  } else if (tid < OW2V) {                // W1vT
    int idx = tid - OW1V; int col = idx >> 5, k = idx & 31;
    v = (k < 10) ? fcvW1[k * 32 + col] : 0.0f;
  } else if (tid < OW1P) {                // W2vT
    int idx = tid - OW2V; int col = idx >> 5, k = idx & 31;
    v = fcvW2[k * 576 + col];
  } else if (tid < OW2PC) {               // W1pT
    int idx = tid - OW1P; int col = idx >> 5, k = idx & 31;
    v = (k < 10) ? fcpW1[k * 64 + col] : 0.0f;
  } else {                                // W2pcT: compact columns for psi
    int idx = tid - OW2PC; int cc = idx >> 6, k = idx & 63;
    if (cc < 48)      v = fcpW2[k * 1152 + (cc / 3) * 32 + (cc % 3)];           // w00[i,o<3]
    else if (cc < 72) { int c2 = cc - 48; v = fcpW2[k * 1152 + 512 + (c2 / 3) * 32 + (c2 % 3)]; } // w11
    else              v = 0.0f;
  }
  wc[tid] = (_Float16)v;
}

// ---------------------------------------------------------------------------
// K1: per-edge geometry: edge_vec, elen, soft-one-hot embedding (f16, padded
// to 16), sh1 (includes sqrt(3)), cutoff.
// ---------------------------------------------------------------------------
__global__ void edge_geom_kernel(const float* __restrict__ pos, const int* __restrict__ esrc,
                                 const int* __restrict__ edst, int E,
                                 _Float16* __restrict__ embh, float* __restrict__ sh1w,
                                 float* __restrict__ cutw) {
  int e = blockIdx.x * blockDim.x + threadIdx.x;
  if (e >= E) return;
  int s = esrc[e], d = edst[e];
  float vx = pos[s * 3 + 0] - pos[d * 3 + 0];
  float vy = pos[s * 3 + 1] - pos[d * 3 + 1];
  float vz = pos[s * 3 + 2] - pos[d * 3 + 2];
  float l = sqrtf(vx * vx + vy * vy + vz * vz + 1e-12f);
  float invl = frcp(l);
  sh1w[e * 3 + 0] = SQRT3 * vx * invl;
  sh1w[e * 3 + 1] = SQRT3 * vy * invl;
  sh1w[e * 3 + 2] = SQRT3 * vz * invl;
  cutw[e] = sus_f(10.0f * (1.0f - l * (1.0f / 3.0f)));
  const float invstep = 11.0f / 3.0f;
#pragma unroll
  for (int j = 0; j < 10; ++j) {
    float vj = 3.0f * (float)(j + 1) / 11.0f;
    float diff = (l - vj) * invstep;
    embh[e * 16 + j] = (_Float16)(EMB_C * sus_f(diff + 1.0f) * sus_f(1.0f - diff));
  }
#pragma unroll
  for (int j = 10; j < 16; ++j) embh[e * 16 + j] = (_Float16)0.0f;
}

// ---------------------------------------------------------------------------
// K2: per-node pre-contracted queries: qt0[j] = sum_i (x0@Wq0/4)[i]*Wd00[i,j],
// qt1[j,d] = sum_i (x1@Wq1/sqrt8)[i,d]*Wd11[i,j].
// ---------------------------------------------------------------------------
__global__ void node_q_kernel(const float* __restrict__ feats, const float* __restrict__ Wq0,
                              const float* __restrict__ Wq1, const float* __restrict__ Wd00,
                              const float* __restrict__ Wd11, int Nn,
                              float* __restrict__ qt0, float* __restrict__ qt1) {
  int n = blockIdx.x * blockDim.x + threadIdx.x;
  if (n >= Nn) return;
  const float* fp = feats + n * 40;
  float x0[16], q0[16];
#pragma unroll
  for (int i = 0; i < 16; ++i) x0[i] = fp[i];
#pragma unroll
  for (int o = 0; o < 16; ++o) {
    float a = 0.0f;
#pragma unroll
    for (int i = 0; i < 16; ++i) a += x0[i] * Wq0[i * 16 + o];
    q0[o] = a * 0.25f;
  }
#pragma unroll
  for (int j = 0; j < 16; ++j) {
    float a = 0.0f;
#pragma unroll
    for (int i = 0; i < 16; ++i) a += q0[i] * Wd00[i * 16 + j];
    qt0[n * 16 + j] = a;
  }
  float x1[8][3], q1[8][3];
#pragma unroll
  for (int i = 0; i < 8; ++i)
#pragma unroll
    for (int d = 0; d < 3; ++d) x1[i][d] = fp[16 + i * 3 + d];
#pragma unroll
  for (int o = 0; o < 8; ++o)
#pragma unroll
    for (int d = 0; d < 3; ++d) {
      float a = 0.0f;
#pragma unroll
      for (int i = 0; i < 8; ++i) a += x1[i][d] * Wq1[i * 8 + o];
      q1[o][d] = a * 0.35355339059327373f;
    }
#pragma unroll
  for (int j = 0; j < 8; ++j)
#pragma unroll
    for (int d = 0; d < 3; ++d) {
      float a = 0.0f;
#pragma unroll
      for (int i = 0; i < 8; ++i) a += q1[i][d] * Wd11[i * 8 + j];
      qt1[n * 24 + j * 3 + d] = a;
    }
}

// ---------------------------------------------------------------------------
// Fused per-tile MLP: w[16x576] = silu(emb[16x32]@W1)[16x32] @ W2[32x576],
// all via v_wmma_f32_16x16x32_f16; result kept in LDS. The 36-chunk loop is
// fully unrolled so the 32B B-column loads batch into s_clauses and the D-tile
// ds_stores hide behind subsequent WMMAs.
// ---------------------------------------------------------------------------
__device__ __forceinline__ void mlp_weights_576(v16h a_emb, int half, int m,
                                                const _Float16* __restrict__ W1T,
                                                const _Float16* __restrict__ W2T,
                                                _Float16* __restrict__ hbuf,
                                                float* __restrict__ wbuf) {
  v8f c0 = {};
  v8f c1 = {};
  v16h b0 = load_b16(W1T + m * 32 + half * 16);
  v16h b1 = load_b16(W1T + (16 + m) * 32 + half * 16);
  c0 = wmma16(a_emb, b0, c0);
  c1 = wmma16(a_emb, b1, c1);
#pragma unroll
  for (int r = 0; r < 8; ++r) {
    int mm = r + 8 * half;            // C/D layout: (M = r+8*half, N = m)
    hbuf[mm * 32 + m]      = (_Float16)silu_f(c0[r]);
    hbuf[mm * 32 + 16 + m] = (_Float16)silu_f(c1[r]);
  }
  __syncthreads();
  v16h a2 = load_a_slice32(hbuf + m * 32, half * 8);
#pragma unroll
  for (int nt = 0; nt < 36; ++nt) {
    v16h b = load_b16(W2T + (nt * 16 + m) * 32 + half * 16);
    v8f acc = {};
    acc = wmma16(a2, b, acc);
#pragma unroll
    for (int r = 0; r < 8; ++r)
      wbuf[(r + 8 * half) * 576 + nt * 16 + m] = acc[r];
  }
  __syncthreads();
}

// edge_tp contraction against a 576-float per-edge weight row in LDS.
// half 0 -> out0[0..7], out1[0..3]; half 1 -> out0[8..15], out1[4..7].
__device__ __forceinline__ void edge_tp_576(const float* __restrict__ wl, int half,
                                            const float x0r[16], const float x1r[8][3],
                                            const float x1s[8], const float sh[3],
                                            float out0[8], float out1[4][3]) {
#pragma unroll
  for (int oo = 0; oo < 8; ++oo) {
    int o = half * 8 + oo;
    float acc = 0.0f;
#pragma unroll
    for (int i = 0; i < 16; ++i) acc += x0r[i] * wl[i * 16 + o];          // w00
#pragma unroll
    for (int i = 0; i < 8; ++i)  acc += x1s[i] * wl[256 + i * 16 + o];    // w11
    out0[oo] = acc * NORM24;
  }
#pragma unroll
  for (int oo = 0; oo < 4; ++oo) {
    int o = half * 4 + oo;
    float s = 0.0f;
#pragma unroll
    for (int i = 0; i < 16; ++i) s += x0r[i] * wl[384 + i * 8 + o];       // w01
#pragma unroll
    for (int d = 0; d < 3; ++d) {
      float acc = s * sh[d] * INV_SQRT3;
#pragma unroll
      for (int i = 0; i < 8; ++i) acc += x1r[i][d] * wl[512 + i * 8 + o]; // w10
      out1[oo][d] = acc * NORM24;
    }
  }
}

// ---------------------------------------------------------------------------
// K3: one wave per 16-edge tile. k path -> dot/expv/z; v path -> vbuf.
// LDS = 38 KB -> 8 waves resident per WGP.
// ---------------------------------------------------------------------------
__global__ void __launch_bounds__(32) edge_kv_kernel(
    const _Float16* __restrict__ embh, const float* __restrict__ sh1w,
    const float* __restrict__ cutw, const float* __restrict__ feats,
    const int* __restrict__ esrc, const int* __restrict__ edst,
    const _Float16* __restrict__ W1kT, const _Float16* __restrict__ W2kT,
    const _Float16* __restrict__ W1vT, const _Float16* __restrict__ W2vT,
    const float* __restrict__ qt0, const float* __restrict__ qt1,
    float* __restrict__ expv, float* __restrict__ zbuf, float* __restrict__ vbuf) {
  __shared__ float    wbuf[16 * 576];   // 36 KB: per-edge MLP weight rows
  __shared__ _Float16 hbuf[16 * 32];    // hidden activations (f16)
  const int lane = threadIdx.x;
  const int half = lane >> 4;
  const int m    = lane & 15;
  const int e    = blockIdx.x * 16 + m;
  const int src  = esrc[e];
  const int dst  = edst[e];
  const float cut = cutw[e];
  float sh[3];
#pragma unroll
  for (int d = 0; d < 3; ++d) sh[d] = sh1w[e * 3 + d];

  float x0r[16], x1r[8][3], x1s[8];
  const float* fp = feats + src * 40;
#pragma unroll
  for (int i = 0; i < 16; ++i) x0r[i] = fp[i];
#pragma unroll
  for (int i = 0; i < 8; ++i) {
#pragma unroll
    for (int d = 0; d < 3; ++d) x1r[i][d] = fp[16 + i * 3 + d];
    x1s[i] = (x1r[i][0] * sh[0] + x1r[i][1] * sh[1] + x1r[i][2] * sh[2]) * INV_SQRT3;
  }

  v16h a_emb = load_a_emb(embh + e * 16, half);

  // ---- k path: weights, tensor product, attention logit ----
  mlp_weights_576(a_emb, half, m, W1kT, W2kT, hbuf, wbuf);
  float k0v[8], k1v[4][3];
  edge_tp_576(wbuf + m * 576, half, x0r, x1r, x1s, sh, k0v, k1v);

  float dp = 0.0f;
  {
    const float* q0p = qt0 + dst * 16 + half * 8;
    const float* q1p = qt1 + dst * 24 + half * 12;
#pragma unroll
    for (int oo = 0; oo < 8; ++oo) dp += k0v[oo] * q0p[oo];
#pragma unroll
    for (int oo = 0; oo < 4; ++oo)
#pragma unroll
      for (int d = 0; d < 3; ++d) dp += INV_SQRT3 * k1v[oo][d] * q1p[oo * 3 + d];
  }
  dp += __shfl_xor(dp, 16, 32);       // combine the two output halves
  if (half == 0) {
    float ev = cut * fexp(dp * NORM24);
    expv[e] = ev;
    atomicAdd(&zbuf[dst], ev);
  }
  __syncthreads();

  // ---- v path ----
  mlp_weights_576(a_emb, half, m, W1vT, W2vT, hbuf, wbuf);
  float v0v[8], v1v[4][3];
  edge_tp_576(wbuf + m * 576, half, x0r, x1r, x1s, sh, v0v, v1v);
  float* vp = vbuf + e * 40;
#pragma unroll
  for (int oo = 0; oo < 8; ++oo) vp[half * 8 + oo] = v0v[oo];
#pragma unroll
  for (int oo = 0; oo < 4; ++oo)
#pragma unroll
    for (int d = 0; d < 3; ++d) vp[16 + (half * 4 + oo) * 3 + d] = v1v[oo][d];
}

// ---------------------------------------------------------------------------
// K4: coef = sqrt(expv / z[dst]); scatter coef * v into att via f32 atomics.
// ---------------------------------------------------------------------------
__global__ void att_accum_kernel(const float* __restrict__ expv, const float* __restrict__ zbuf,
                                 const float* __restrict__ vbuf, const int* __restrict__ edst,
                                 int E, float* __restrict__ attw) {
  int e = blockIdx.x * blockDim.x + threadIdx.x;
  if (e >= E) return;
  int dst = edst[e];
  float zz = zbuf[dst];
  zz = (zz == 0.0f) ? 1.0f : zz;
  float alpha = expv[e] * frcp(zz);
  float coef = (alpha > 0.0f) ? sqrtf(alpha) : 0.0f;
  const float* vp = vbuf + e * 40;
  float* ap = attw + dst * 40;
#pragma unroll
  for (int j = 0; j < 40; ++j) atomicAdd(&ap[j], coef * vp[j]);
}

// ---------------------------------------------------------------------------
// K5: pruned p tensor product -> psi (only pot[:, :3] is ever used).
// One wave per 16-edge tile; 14 WMMAs against the compact 64x80 fcp_W2 slice.
// ---------------------------------------------------------------------------
__global__ void __launch_bounds__(32) edge_p_kernel(
    const _Float16* __restrict__ embh, const float* __restrict__ sh1w,
    const float* __restrict__ cutw, const float* __restrict__ attw,
    const int* __restrict__ esrc, const int* __restrict__ edst,
    const _Float16* __restrict__ W1pT, const _Float16* __restrict__ W2pcT,
    float* __restrict__ psi) {
  __shared__ _Float16 hb[16 * 64];
  __shared__ float    wc[16 * 80];
  const int lane = threadIdx.x;
  const int half = lane >> 4;
  const int m    = lane & 15;
  const int e    = blockIdx.x * 16 + m;

  v16h a = load_a_emb(embh + e * 16, half);
#pragma unroll
  for (int nt = 0; nt < 4; ++nt) {                 // hidden = 64 -> 4 N-chunks
    v16h b = load_b16(W1pT + (nt * 16 + m) * 32 + half * 16);
    v8f acc = {};
    acc = wmma16(a, b, acc);
#pragma unroll
    for (int r = 0; r < 8; ++r)
      hb[(r + 8 * half) * 64 + nt * 16 + m] = (_Float16)silu_f(acc[r]);
  }
  __syncthreads();
  v16h alo = load_a_slice32(hb + m * 64, half * 8);        // K 0..31
  v16h ahi = load_a_slice32(hb + m * 64 + 32, half * 8);   // K 32..63
#pragma unroll
  for (int nt = 0; nt < 5; ++nt) {                 // 80 compact cols -> 5 chunks
    v16h blo = load_b16(W2pcT + (nt * 16 + m) * 64 + half * 16);
    v16h bhi = load_b16(W2pcT + (nt * 16 + m) * 64 + 32 + half * 16);
    v8f acc = {};
    acc = wmma16(alo, blo, acc);
    acc = wmma16(ahi, bhi, acc);
#pragma unroll
    for (int r = 0; r < 8; ++r)
      wc[(r + 8 * half) * 80 + nt * 16 + m] = acc[r];
  }
  __syncthreads();

  const int src = esrc[e], dst = edst[e];
  const float cut = cutw[e];
  float sh[3];
#pragma unroll
  for (int d = 0; d < 3; ++d) sh[d] = sh1w[e * 3 + d];
  const float* ap = attw + src * 40;

  if (half == 0) {                                 // w00 part: a0 . wc[:, i*3+o]
    float a0[16];
#pragma unroll
    for (int i = 0; i < 16; ++i) a0[i] = ap[i];
#pragma unroll
    for (int o = 0; o < 3; ++o) {
      float s = 0.0f;
#pragma unroll
      for (int i = 0; i < 16; ++i) s += a0[i] * wc[m * 80 + i * 3 + o];
      atomicAdd(&psi[dst * 3 + o], cut * NORM24 * s);
    }
  } else {                                         // w11 part: a1s . wc[:, 48+i*3+o]
    float a1s[8];
#pragma unroll
    for (int i = 0; i < 8; ++i) {
      float v0 = ap[16 + i * 3], v1 = ap[16 + i * 3 + 1], v2 = ap[16 + i * 3 + 2];
      a1s[i] = (v0 * sh[0] + v1 * sh[1] + v2 * sh[2]) * INV_SQRT3;
    }
#pragma unroll
    for (int o = 0; o < 3; ++o) {
      float s = 0.0f;
#pragma unroll
      for (int i = 0; i < 8; ++i) s += a1s[i] * wc[m * 80 + 48 + i * 3 + o];
      atomicAdd(&psi[dst * 3 + o], cut * NORM24 * s);
    }
  }
}

// ---------------------------------------------------------------------------
// K6: att[:, :3] += 0.1 * curl(psi); write final output.
// ---------------------------------------------------------------------------
__global__ void finalize_kernel(const float* __restrict__ attw, const float* __restrict__ psi,
                                int Nn, float* __restrict__ out) {
  int tid = blockIdx.x * blockDim.x + threadIdx.x;
  if (tid >= Nn * 40) return;
  int n = tid / 40;
  int j = tid - n * 40;
  float v = attw[tid];
  if (j < 3) {
    float p0 = psi[n * 3 + 0], p1 = psi[n * 3 + 1], p2 = psi[n * 3 + 2];
    float c = (j == 0) ? (p2 - p1) : (j == 1) ? (p0 - p2) : (p1 - p0);
    v += 0.1f * c;
  }
  out[tid] = v;
}

// ---------------------------------------------------------------------------
extern "C" void kernel_launch(void* const* d_in, const int* in_sizes, int n_in,
                              void* d_out, int out_size, void* d_ws, size_t ws_size,
                              hipStream_t stream) {
  const float* feats = (const float*)d_in[0];
  const float* pos   = (const float*)d_in[1];
  const int*   esrc  = (const int*)d_in[2];
  const int*   edst  = (const int*)d_in[3];
  const float* Wq0   = (const float*)d_in[4];
  const float* Wq1   = (const float*)d_in[5];
  const float* fckW1 = (const float*)d_in[6];
  const float* fckW2 = (const float*)d_in[7];
  const float* fcvW1 = (const float*)d_in[8];
  const float* fcvW2 = (const float*)d_in[9];
  const float* Wd00  = (const float*)d_in[10];
  const float* Wd11  = (const float*)d_in[11];
  const float* fcpW1 = (const float*)d_in[12];
  const float* fcpW2 = (const float*)d_in[13];

  const int E  = in_sizes[2];       // 98304 (multiple of 16)
  const int Nn = in_sizes[1] / 3;   // 8192

  char* base = (char*)d_ws;
  size_t off = 0;
  auto take = [&](size_t bytes) -> void* {
    void* p = base + off;
    off += (bytes + 255) & ~(size_t)255;
    return p;
  };
  _Float16* embh = (_Float16*)take((size_t)E * 16 * 2);
  float* sh1w    = (float*)take((size_t)E * 3 * 4);
  float* cutw    = (float*)take((size_t)E * 4);
  float* qt0     = (float*)take((size_t)Nn * 16 * 4);
  float* qt1     = (float*)take((size_t)Nn * 24 * 4);
  float* expv    = (float*)take((size_t)E * 4);
  float* zbuf    = (float*)take((size_t)Nn * 4);
  float* vbuf    = (float*)take((size_t)E * 40 * 4);
  float* attw    = (float*)take((size_t)Nn * 40 * 4);
  float* psi     = (float*)take((size_t)Nn * 3 * 4);
  _Float16* wcst = (_Float16*)take((size_t)WTOT * 2);
  _Float16* W1kT  = wcst + OW1K;
  _Float16* W2kT  = wcst + OW2K;
  _Float16* W1vT  = wcst + OW1V;
  _Float16* W2vT  = wcst + OW2V;
  _Float16* W1pT  = wcst + OW1P;
  _Float16* W2pcT = wcst + OW2PC;

  hipMemsetAsync(zbuf, 0, (size_t)Nn * 4, stream);
  hipMemsetAsync(attw, 0, (size_t)Nn * 40 * 4, stream);
  hipMemsetAsync(psi,  0, (size_t)Nn * 3 * 4, stream);

  prep_weights_kernel<<<(WTOT + 255) / 256, 256, 0, stream>>>(fckW1, fckW2, fcvW1, fcvW2,
                                                              fcpW1, fcpW2, wcst);
  edge_geom_kernel<<<(E + 255) / 256, 256, 0, stream>>>(pos, esrc, edst, E, embh, sh1w, cutw);
  node_q_kernel<<<(Nn + 255) / 256, 256, 0, stream>>>(feats, Wq0, Wq1, Wd00, Wd11, Nn, qt0, qt1);
  edge_kv_kernel<<<E / 16, 32, 0, stream>>>(embh, sh1w, cutw, feats, esrc, edst,
                                            W1kT, W2kT, W1vT, W2vT, qt0, qt1,
                                            expv, zbuf, vbuf);
  att_accum_kernel<<<(E + 255) / 256, 256, 0, stream>>>(expv, zbuf, vbuf, edst, E, attw);
  edge_p_kernel<<<E / 16, 32, 0, stream>>>(embh, sh1w, cutw, attw, esrc, edst,
                                           W1pT, W2pcT, psi);
  finalize_kernel<<<(Nn * 40 + 255) / 256, 256, 0, stream>>>(attw, psi, Nn, (float*)d_out);

  (void)n_in; (void)out_size; (void)ws_size;
}
